// SLA2Attention_24129126268934
// MI455X (gfx1250) — matrix operand order repaired and checked by reading, not compiled, not run on hardware
//
#include <hip/hip_runtime.h>
#include <hip/hip_bf16.h>
#include <math.h>
#include <stdint.h>

typedef __attribute__((ext_vector_type(16))) _Float16 v16h;
typedef __attribute__((ext_vector_type(8)))  _Float16 v8h;
typedef __attribute__((ext_vector_type(4)))  _Float16 v4h;
typedef __attribute__((ext_vector_type(8)))  float    v8f;

// Problem constants (SLA2 reference)
#define BB    2
#define NN    1024
#define CC    1024
#define HH    16
#define DD    64
#define BLKC  8
#define NBC   128
#define TOPKC 7
#define EPSC  1e-6f
#define SCALEC 0.125f   // D^-0.5

// ---------------- wave32 reductions ----------------
__device__ __forceinline__ float wave_sum(float v) {
    #pragma unroll
    for (int o = 16; o > 0; o >>= 1) v += __shfl_xor(v, o);
    return v;
}
__device__ __forceinline__ float wave_max(float v) {
    #pragma unroll
    for (int o = 16; o > 0; o >>= 1) v = fmaxf(v, __shfl_xor(v, o));
    return v;
}

__device__ __forceinline__ v16h ld_frag(const _Float16* lo, const _Float16* hi) {
    v8h a = *(const v8h*)lo;
    v8h b = *(const v8h*)hi;
    v16h r;
    #pragma unroll
    for (int e = 0; e < 8; ++e) { r[e] = a[e]; r[e + 8] = b[e]; }
    return r;
}

// CDNA5 async global->LDS copy of 16 bytes per lane (ASYNCcnt-tracked).
// LDS byte offset = low 32 bits of the flat shared-aperture address.
__device__ __forceinline__ void async_copy_b128(const void* lds_dst, const void* gsrc) {
    uint32_t loff = (uint32_t)(uintptr_t)lds_dst;
    uint64_t gaddr = (uint64_t)(uintptr_t)gsrc;
    asm volatile("global_load_async_to_lds_b128 %0, %1, off"
                 :: "v"(loff), "v"(gaddr) : "memory");
}
__device__ __forceinline__ void wait_async0() {
    asm volatile("s_wait_asynccnt 0x0" ::: "memory");
}
// Async loads complete in order: "<= 3 outstanding" means the oldest tile
// (3 async ops per wave per tile) has fully landed while the newest flies.
__device__ __forceinline__ void wait_async3() {
    asm volatile("s_wait_asynccnt 0x3" ::: "memory");
}

// ---------------- f32 -> f16 conversion ----------------
__global__ __launch_bounds__(256)
void cvt_f16_kernel(const float* __restrict__ src, _Float16* __restrict__ dst, size_t n) {
    size_t i = (size_t)blockIdx.x * blockDim.x + threadIdx.x;
    size_t stride = (size_t)gridDim.x * blockDim.x;
    for (; i < n; i += stride) dst[i] = (_Float16)src[i];
}

// ---------------- big WMMA f16 GEMM: async DMA + triple buffering ----------
// C[M x Nc] = A[M x K] @ W[K x Nc] (+ bias[Nc]); row-major; f32 out.
// WG = 256 threads = 8 waves arranged 2(M) x 4(N); wave tile 32x32 (4 WMMA);
// WG tile 64(M) x 128(N); K-step 32; 3-deep LDS pipeline fed by
// global_load_async_to_lds_b128 with partial s_wait_asynccnt waits.
// Requires M%64==0, Nc%128==0, K%32==0.
__global__ __launch_bounds__(256)
void wmma_gemm_async(const _Float16* __restrict__ A, const _Float16* __restrict__ W,
                     float* __restrict__ Cm, const float* __restrict__ bias,
                     int M, int Nc, int K) {
    __shared__ __align__(16) _Float16 As[3][64][40];    // 64 M-rows x 32 K (padded)
    __shared__ __align__(16) _Float16 Ws[3][32][136];   // 32 K-rows x 128 N (padded)

    const int t    = threadIdx.x;
    const int lane = t & 31;
    const int wave = t >> 5;
    const int wm   = wave >> 2;     // 0..1 : M sub-block of 32
    const int wn   = wave & 3;      // 0..3 : N sub-block of 32

    const int rowA0 = blockIdx.y * 64;
    const int colW0 = blockIdx.x * 128;

    // async chunk assignment (16B chunks)
    const int a_row  = t >> 2,      a_ch = (t & 3) * 8;    // A: 64r x 4 chunks
    const int w_row0 = t >> 4,      w_ch = (t & 15) * 8;   // W: 32r x 16 chunks, 2/thread
    const int w_row1 = 16 + (t >> 4);

    // per-thread streaming pointers, advanced by one K-step per issued tile
    const _Float16* gA  = A + (size_t)(rowA0 + a_row) * K + a_ch;
    const _Float16* gW0 = W + (size_t)w_row0 * Nc + colW0 + w_ch;
    const _Float16* gW1 = W + (size_t)w_row1 * Nc + colW0 + w_ch;
    const size_t wStep = (size_t)32 * Nc;

    auto issue_tile = [&](int buf) {
        async_copy_b128(&As[buf][a_row][a_ch],  gA);
        async_copy_b128(&Ws[buf][w_row0][w_ch], gW0);
        async_copy_b128(&Ws[buf][w_row1][w_ch], gW1);
        gA += 32; gW0 += wStep; gW1 += wStep;
    };

    v8f acc00 = {}, acc01 = {}, acc10 = {}, acc11 = {};

    const int nsteps = K >> 5;
    issue_tile(0);
    if (nsteps > 1) issue_tile(1);

    int cur = 0;
    for (int i = 0; i < nsteps; ++i) {
        if (i + 1 < nsteps) wait_async3();   // oldest tile landed, next still in flight
        else                wait_async0();
        __syncthreads();                     // all waves' DMA done; prev reads finished
        if (i + 2 < nsteps) issue_tile((cur + 2 >= 3) ? cur - 1 : cur + 2);

        // fragments (A: lane<16 -> K {0..7,16..23}, lane>=16 -> K {8..15,24..31};
        //            B: lane%16 = N column, same K order as A)
        const int klo = (lane >> 4) * 8;
        const int r0  = wm * 32 + (lane & 15);
        const int r1  = r0 + 16;
        const int nB  = wn * 32;
        v16h a0 = ld_frag(&As[cur][r0][klo], &As[cur][r0][klo + 16]);
        v16h a1 = ld_frag(&As[cur][r1][klo], &As[cur][r1][klo + 16]);
        v16h b0, b1;
        #pragma unroll
        for (int e = 0; e < 8; ++e) {
            b0[e]     = Ws[cur][klo + e][nB + (lane & 15)];
            b0[e + 8] = Ws[cur][klo + 16 + e][nB + (lane & 15)];
            b1[e]     = Ws[cur][klo + e][nB + 16 + (lane & 15)];
            b1[e + 8] = Ws[cur][klo + 16 + e][nB + 16 + (lane & 15)];
        }

        acc00 = __builtin_amdgcn_wmma_f32_16x16x32_f16(false, a0, false, b0, (short)0, acc00, false, false);
        acc01 = __builtin_amdgcn_wmma_f32_16x16x32_f16(false, a0, false, b1, (short)0, acc01, false, false);
        acc10 = __builtin_amdgcn_wmma_f32_16x16x32_f16(false, a1, false, b0, (short)0, acc10, false, false);
        acc11 = __builtin_amdgcn_wmma_f32_16x16x32_f16(false, a1, false, b1, (short)0, acc11, false, false);

        cur = (cur == 2) ? 0 : cur + 1;
    }

    // C/D layout: VGPR r -> (M=r, N=lane) lanes 0-15, (M=r+8, N=lane-16) lanes 16-31.
    const int nc0 = colW0 + wn * 32 + (lane & 15);
    const int nc1 = nc0 + 16;
    const int mb0 = rowA0 + wm * 32 + ((lane < 16) ? 0 : 8);
    const int mb1 = mb0 + 16;
    const float bd0 = bias ? bias[nc0] : 0.0f;
    const float bd1 = bias ? bias[nc1] : 0.0f;
    #pragma unroll
    for (int r = 0; r < 8; ++r) {
        Cm[(size_t)(mb0 + r) * Nc + nc0] = acc00[r] + bd0;
        Cm[(size_t)(mb0 + r) * Nc + nc1] = acc01[r] + bd1;
        Cm[(size_t)(mb1 + r) * Nc + nc0] = acc10[r] + bd0;
        Cm[(size_t)(mb1 + r) * Nc + nc1] = acc11[r] + bd1;
    }
}

// ---------------- small WMMA f16 GEMM (batched; used for kv = phi_k^T v) ----
// WG = 256 = 8 waves; WG tile 32(M) x 64(N); wave tile 16x16.
__global__ __launch_bounds__(256)
void wmma_gemm_f16(const _Float16* __restrict__ A, const _Float16* __restrict__ W,
                   float* __restrict__ Cm, const float* __restrict__ bias,
                   int M, int Nc, int K,
                   long long sAb, long long sWb, long long sCb) {
    __shared__ __align__(16) _Float16 As[32][40];
    __shared__ __align__(16) _Float16 Ws[32][72];

    const int t    = threadIdx.x;
    const int lane = t & 31;
    const int wave = t >> 5;
    const int wm   = wave >> 2;
    const int wn   = wave & 3;

    const _Float16* Ab = A + (size_t)blockIdx.z * (size_t)sAb;
    const _Float16* Wb = W + (size_t)blockIdx.z * (size_t)sWb;
    float*          Cb = Cm + (size_t)blockIdx.z * (size_t)sCb;

    const int rowA0 = blockIdx.y * 32;
    const int colW0 = blockIdx.x * 64;
    const int ar = t >> 3, ac = (t & 7) * 4;
    const int wr = t >> 3, wc = (t & 7) * 8;

    v8f acc = {};
    for (int k0 = 0; k0 < K; k0 += 32) {
        *(v4h*)&As[ar][ac] = *(const v4h*)(Ab + (size_t)(rowA0 + ar) * K + (k0 + ac));
        *(v8h*)&Ws[wr][wc] = *(const v8h*)(Wb + (size_t)(k0 + wr) * Nc + (colW0 + wc));
        __syncthreads();

        const int m   = wm * 16 + (lane & 15);
        const int klo = (lane >> 4) * 8;
        const int n0  = wn * 16;
        v16h af = ld_frag(&As[m][klo], &As[m][klo + 16]);
        v16h bf;
        #pragma unroll
        for (int e = 0; e < 8; ++e) {
            bf[e]     = Ws[klo + e][n0 + (lane & 15)];
            bf[e + 8] = Ws[klo + 16 + e][n0 + (lane & 15)];
        }
        acc = __builtin_amdgcn_wmma_f32_16x16x32_f16(false, af, false, bf, (short)0, acc, false, false);
        __syncthreads();
    }

    const int nCol  = colW0 + wn * 16 + (lane & 15);
    const int mBase = rowA0 + wm * 16 + ((lane < 16) ? 0 : 8);
    const float badd = bias ? bias[nCol] : 0.0f;
    #pragma unroll
    for (int r = 0; r < 8; ++r)
        Cb[(size_t)(mBase + r) * Nc + nCol] = acc[r] + badd;
}

// ---------------- LayerNorm(qk) + phi softmax + v->f16 ----------------
__global__ __launch_bounds__(256)
void ln_phi_kernel(const float* __restrict__ qkv,
                   const float* __restrict__ qw, const float* __restrict__ qb,
                   const float* __restrict__ kw, const float* __restrict__ kb,
                   float* __restrict__ qf, float* __restrict__ kf,
                   float* __restrict__ phiq, _Float16* __restrict__ vh,
                   _Float16* __restrict__ phikT) {
    const int lane = threadIdx.x & 31;
    const int wave = threadIdx.x >> 5;
    const int qidx = blockIdx.x * 8 + wave;         // (b*H + h)*N + n
    const int b  = qidx / (HH * NN);
    const int h  = (qidx / NN) % HH;
    const int n  = qidx % NN;
    const int bh = qidx / NN;
    const size_t row = (size_t)(b * NN + n) * (3 * CC);
    const int d0 = lane, d1 = lane + 32;

    // ---- q ----
    {
        float x0 = qkv[row + h * DD + d0];
        float x1 = qkv[row + h * DD + d1];
        float m  = wave_sum(x0 + x1) * (1.0f / 64.0f);
        float va = wave_sum((x0 - m) * (x0 - m) + (x1 - m) * (x1 - m)) * (1.0f / 64.0f);
        float inv = rsqrtf(va + EPSC);
        float y0 = (x0 - m) * inv * qw[d0] + qb[d0];
        float y1 = (x1 - m) * inv * qw[d1] + qb[d1];
        qf[(size_t)qidx * DD + d0] = y0;
        qf[(size_t)qidx * DD + d1] = y1;
        float mx = wave_max(fmaxf(y0, y1));
        float e0 = expf(y0 - mx), e1 = expf(y1 - mx);
        float s  = wave_sum(e0 + e1);
        phiq[(size_t)qidx * DD + d0] = e0 / s;
        phiq[(size_t)qidx * DD + d1] = e1 / s;
    }
    // ---- k ----
    {
        float x0 = qkv[row + CC + h * DD + d0];
        float x1 = qkv[row + CC + h * DD + d1];
        float m  = wave_sum(x0 + x1) * (1.0f / 64.0f);
        float va = wave_sum((x0 - m) * (x0 - m) + (x1 - m) * (x1 - m)) * (1.0f / 64.0f);
        float inv = rsqrtf(va + EPSC);
        float y0 = (x0 - m) * inv * kw[d0] + kb[d0];
        float y1 = (x1 - m) * inv * kw[d1] + kb[d1];
        kf[(size_t)qidx * DD + d0] = y0;
        kf[(size_t)qidx * DD + d1] = y1;
        float mx = wave_max(fmaxf(y0, y1));
        float e0 = expf(y0 - mx), e1 = expf(y1 - mx);
        float s  = wave_sum(e0 + e1);
        phikT[((size_t)bh * DD + d0) * NN + n] = (_Float16)(e0 / s);
        phikT[((size_t)bh * DD + d1) * NN + n] = (_Float16)(e1 / s);
    }
    // ---- v -> f16 ----
    vh[(size_t)qidx * DD + d0] = (_Float16)qkv[row + 2 * CC + h * DD + d0];
    vh[(size_t)qidx * DD + d1] = (_Float16)qkv[row + 2 * CC + h * DD + d1];
}

// ---------------- mean-pool k over blocks of 8 ----------------
__global__ __launch_bounds__(256)
void kcmp_kernel(const float* __restrict__ kf, float* __restrict__ kcmp) {
    int idx = blockIdx.x * blockDim.x + threadIdx.x;   // (bh*NB + nb)*64 + d
    int d   = idx & 63;
    int nb  = (idx >> 6) & (NBC - 1);
    int bh  = idx >> (6 + 7);
    float s = 0.0f;
    #pragma unroll
    for (int j = 0; j < BLKC; ++j)
        s += kf[((size_t)bh * NN + nb * BLKC + j) * DD + d];
    kcmp[idx] = s * (1.0f / BLKC);
}

// ---------------- z[b,h,d] = sum_n phi_k[n,d] ----------------
__global__ __launch_bounds__(256)
void zsum_kernel(const _Float16* __restrict__ phikT, float* __restrict__ zv) {
    int idx = blockIdx.x * blockDim.x + threadIdx.x;   // bh*64 + d
    const _Float16* p = phikT + (size_t)idx * NN;
    float s = 0.0f;
    for (int n = 0; n < NN; ++n) s += (float)p[n];
    zv[idx] = s;
}

// ---------------- router + top-7 (lowest-index tie-break) ----------------
__global__ __launch_bounds__(256)
void router_topk_kernel(const float* __restrict__ qf, const float* __restrict__ kcmp,
                        int* __restrict__ topk) {
    const int lane = threadIdx.x & 31;
    const int wave = threadIdx.x >> 5;
    const int qidx = blockIdx.x * 8 + wave;
    const int bh   = qidx / NN;
    const int nbb  = lane * 4;

    float r[4] = {0.f, 0.f, 0.f, 0.f};
    for (int d = 0; d < DD; ++d) {
        float qd = qf[(size_t)qidx * DD + d] * SCALEC;
        #pragma unroll
        for (int j = 0; j < 4; ++j)
            r[j] += qd * kcmp[((size_t)bh * NBC + nbb + j) * DD + d];
    }
    #pragma unroll
    for (int t = 0; t < TOPKC; ++t) {
        float lv = -3.0e38f; int li = 0x7fffffff;
        #pragma unroll
        for (int j = 0; j < 4; ++j)
            if (r[j] > lv) { lv = r[j]; li = nbb + j; }
        float bv = lv; int bi = li;
        #pragma unroll
        for (int o = 16; o > 0; o >>= 1) {
            float ov = __shfl_xor(bv, o);
            int   oi = __shfl_xor(bi, o);
            if (ov > bv || (ov == bv && oi < bi)) { bv = ov; bi = oi; }
        }
        if ((bi >> 2) == lane) r[bi & 3] = -3.0e38f;
        if (lane == 0) topk[(size_t)qidx * 8 + t] = bi;
    }
}

// ---------------- sparse softmax attn (56 keys) + linear branch + combine ----
__global__ __launch_bounds__(256)
void sparse_linear_kernel(const float* __restrict__ qf, const float* __restrict__ kf,
                          const float* __restrict__ phiq, const _Float16* __restrict__ vh,
                          const float* __restrict__ kv, const float* __restrict__ zv,
                          const int* __restrict__ topk, _Float16* __restrict__ comb) {
    __shared__ float qs[8][64];
    __shared__ float fs[8][64];
    __shared__ float ps[8][64];
    __shared__ int   nk[8][64];

    const int lane = threadIdx.x & 31;
    const int wave = threadIdx.x >> 5;
    const int qidx = blockIdx.x * 8 + wave;
    const int b  = qidx / (HH * NN);
    const int h  = (qidx / NN) % HH;
    const int n  = qidx % NN;
    const int bh = qidx / NN;

    qs[wave][lane]      = qf[(size_t)qidx * DD + lane];
    qs[wave][lane + 32] = qf[(size_t)qidx * DD + lane + 32];
    fs[wave][lane]      = phiq[(size_t)qidx * DD + lane];
    fs[wave][lane + 32] = phiq[(size_t)qidx * DD + lane + 32];
    __syncthreads();

    float lg[2]; int nn[2];
    #pragma unroll
    for (int i = 0; i < 2; ++i) {
        int s = lane + 32 * i;
        int t = s >> 3;
        if (t < TOPKC) {
            int blk = topk[(size_t)qidx * 8 + t];
            int nkey = blk * BLKC + (s & 7);
            const float* kp = kf + ((size_t)bh * NN + nkey) * DD;
            float acc = 0.0f;
            #pragma unroll
            for (int d = 0; d < DD; ++d) acc += qs[wave][d] * kp[d];
            lg[i] = acc * SCALEC;
            nn[i] = nkey;
        } else { lg[i] = -3.0e38f; nn[i] = 0; }
    }
    float mx = wave_max(fmaxf(lg[0], lg[1]));
    float e0 = expf(lg[0] - mx), e1 = expf(lg[1] - mx);
    float inv = 1.0f / wave_sum(e0 + e1);
    ps[wave][lane]      = e0 * inv;
    ps[wave][lane + 32] = e1 * inv;
    nk[wave][lane]      = nn[0];
    nk[wave][lane + 32] = nn[1];

    float dpart = fs[wave][lane] * zv[(size_t)bh * DD + lane]
                + fs[wave][lane + 32] * zv[(size_t)bh * DD + lane + 32];
    float denom = wave_sum(dpart) + EPSC;
    __syncthreads();

    #pragma unroll
    for (int i = 0; i < 2; ++i) {
        int d = lane + 32 * i;
        float accS = 0.0f;
        for (int s = 0; s < TOPKC * BLKC; ++s)
            accS += ps[wave][s] * (float)vh[((size_t)bh * NN + nk[wave][s]) * DD + d];
        float accL = 0.0f;
        #pragma unroll
        for (int dd = 0; dd < DD; ++dd)
            accL += fs[wave][dd] * kv[((size_t)bh * DD + dd) * DD + d];
        comb[(size_t)(b * NN + n) * CC + h * DD + d] = (_Float16)(accS + accL / denom);
    }
}

// ---------------- host-side pipeline ----------------
extern "C" void kernel_launch(void* const* d_in, const int* in_sizes, int n_in,
                              void* d_out, int out_size, void* d_ws, size_t ws_size,
                              hipStream_t stream) {
    const float* x      = (const float*)d_in[0];
    const float* w_qkv  = (const float*)d_in[1];
    const float* b_qkv  = (const float*)d_in[2];
    const float* q_nw   = (const float*)d_in[3];
    const float* q_nb   = (const float*)d_in[4];
    const float* k_nw   = (const float*)d_in[5];
    const float* k_nb   = (const float*)d_in[6];
    const float* w_proj = (const float*)d_in[7];
    const float* b_proj = (const float*)d_in[8];
    float* out = (float*)d_out;

    const size_t BN = (size_t)BB * NN;                // 2048
    char* ws = (char*)d_ws;
    size_t off = 0;
    auto alloc = [&](size_t bytes) -> char* {
        char* p = ws + off;
        off += (bytes + 255) & ~(size_t)255;
        return p;
    };
    float*    qkv32  = (float*)   alloc(BN * 3 * CC * sizeof(float));
    _Float16* xh     = (_Float16*)alloc(BN * CC * sizeof(_Float16));
    _Float16* wqkvh  = (_Float16*)alloc((size_t)CC * 3 * CC * sizeof(_Float16));
    _Float16* wprojh = (_Float16*)alloc((size_t)CC * CC * sizeof(_Float16));
    float*    qf     = (float*)   alloc((size_t)BB * HH * NN * DD * sizeof(float));
    float*    kf     = (float*)   alloc((size_t)BB * HH * NN * DD * sizeof(float));
    float*    phiq   = (float*)   alloc((size_t)BB * HH * NN * DD * sizeof(float));
    _Float16* vh     = (_Float16*)alloc((size_t)BB * HH * NN * DD * sizeof(_Float16));
    _Float16* phikT  = (_Float16*)alloc((size_t)BB * HH * DD * NN * sizeof(_Float16));
    float*    kcmp   = (float*)   alloc((size_t)BB * HH * NBC * DD * sizeof(float));
    float*    zv     = (float*)   alloc((size_t)BB * HH * DD * sizeof(float));
    float*    kvb    = (float*)   alloc((size_t)BB * HH * DD * DD * sizeof(float));
    int*      topk   = (int*)     alloc((size_t)BB * HH * NN * 8 * sizeof(int));
    _Float16* comb   = (_Float16*)alloc(BN * CC * sizeof(_Float16));
    (void)ws_size; (void)in_sizes; (void)n_in; (void)out_size;

    // 1) f32 -> f16 conversions
    cvt_f16_kernel<<<2048, 256, 0, stream>>>(x,      xh,     BN * CC);
    cvt_f16_kernel<<<2048, 256, 0, stream>>>(w_qkv,  wqkvh,  (size_t)CC * 3 * CC);
    cvt_f16_kernel<<<2048, 256, 0, stream>>>(w_proj, wprojh, (size_t)CC * CC);

    // 2) QKV projection: (2048x1024) @ (1024x3072) + b_qkv  [async WMMA]
    wmma_gemm_async<<<dim3(3 * CC / 128, BN / 64, 1), 256, 0, stream>>>(
        xh, wqkvh, qkv32, b_qkv, (int)BN, 3 * CC, CC);

    // 3) LN + qk-norm + phi softmax + v->f16
    ln_phi_kernel<<<(BB * HH * NN) / 8, 256, 0, stream>>>(
        qkv32, q_nw, q_nb, k_nw, k_nb, qf, kf, phiq, vh, phikT);

    // 4) compressed keys + z
    kcmp_kernel<<<(BB * HH * NBC * DD) / 256, 256, 0, stream>>>(kf, kcmp);
    zsum_kernel<<<(BB * HH * DD) / 256, 256, 0, stream>>>(phikT, zv);

    // 5) kv = phi_k^T @ v per (b,h): batched 64x64x1024 GEMM  [WMMA]
    wmma_gemm_f16<<<dim3(1, 2, BB * HH), 256, 0, stream>>>(
        phikT, vh, kvb, nullptr, DD, DD, NN,
        (long long)DD * NN, (long long)NN * DD, (long long)DD * DD);

    // 6) router + top-7
    router_topk_kernel<<<(BB * HH * NN) / 8, 256, 0, stream>>>(qf, kcmp, topk);

    // 7) sparse attention + linear branch + combine -> f16
    sparse_linear_kernel<<<(BB * HH * NN) / 8, 256, 0, stream>>>(
        qf, kf, phiq, vh, kvb, zv, topk, comb);

    // 8) output projection: (2048x1024) @ (1024x1024) + b_proj -> d_out  [async WMMA]
    wmma_gemm_async<<<dim3(CC / 128, BN / 64, 1), 256, 0, stream>>>(
        comb, wprojh, out, b_proj, (int)BN, CC, CC);
}
